// GCNModelWTopK_89541478187031
// MI455X (gfx1250) — compile-verified, matrix operand order checked by reading, not run on hardware
//
#include <hip/hip_runtime.h>
#include <math.h>

#define NG    15135
#define NPAD  15136      // 946 * 16
#define BSZ   8
#define EDG   242160
#define FIN   64
#define HD    128
#define KTOP  7568
#define HFD   512
#define NC    2
#define SORTN 16384

typedef __attribute__((ext_vector_type(16))) _Float16 v16h;
typedef __attribute__((ext_vector_type(8)))  float    v8f;

// ---------------------------------------------------------------- init / norm
__global__ __launch_bounds__(256) void k_init(float* score, float* proj,
                                              float* h512, unsigned* deg) {
  int i = blockIdx.x * 256 + threadIdx.x;
  if (i < BSZ * NG) { score[i] = 0.f; proj[i] = 0.f; }
  if (i < NG) deg[i] = 0u;
  if (i < BSZ * HFD) h512[i] = 0.f;
}

__global__ __launch_bounds__(256) void k_deg(const int* __restrict__ dst,
                                             unsigned* __restrict__ deg) {
  int e = blockIdx.x * 256 + threadIdx.x;
  if (e < EDG) atomicAdd(&deg[dst[e]], 1u);
}

__global__ __launch_bounds__(256) void k_dinv(const unsigned* __restrict__ deg,
                                              float* __restrict__ dinv,
                                              float* __restrict__ selfn) {
  int n = blockIdx.x * 256 + threadIdx.x;
  if (n < NG) {
    float d = rsqrtf((float)(deg[n] + 1u));   // +1 self loop
    dinv[n] = d;
    selfn[n] = d * d;
  }
}

__global__ __launch_bounds__(256) void k_enorm(const int* __restrict__ src,
                                               const int* __restrict__ dst,
                                               const float* __restrict__ dinv,
                                               float* __restrict__ en) {
  int e = blockIdx.x * 256 + threadIdx.x;
  if (e < EDG) en[e] = dinv[src[e]] * dinv[dst[e]];
}

__global__ __launch_bounds__(256) void k_invnorm(const float* __restrict__ w,
                                                 float* __restrict__ out) {
  __shared__ float red[256];
  float s = 0.f;
  for (int i = threadIdx.x; i < 3 * HD; i += 256) s += w[i] * w[i];
  red[threadIdx.x] = s; __syncthreads();
  for (int o = 128; o; o >>= 1) {
    if (threadIdx.x < o) red[threadIdx.x] += red[threadIdx.x + o];
    __syncthreads();
  }
  if (threadIdx.x == 0) out[0] = rsqrtf(red[0]);
}

// ---------------------------------------------------------------- conversions
__global__ __launch_bounds__(256) void k_convert_x(const float* __restrict__ x,
                                                   _Float16* __restrict__ xf) {
  int i = blockIdx.x * 256 + threadIdx.x;          // [b][npad][64]
  if (i >= BSZ * NPAD * FIN) return;
  int f = i & (FIN - 1);
  int nb = i / FIN;
  int n = nb % NPAD, b = nb / NPAD;
  _Float16 v = (_Float16)0.f;
  if (n < NG) v = (_Float16)x[((size_t)b * NG + n) * FIN + f];
  xf[i] = v;
}

// pack W[K,128] f32 into K-pair dwords: Wp[k/2][c] = (h(W[k][c]), h(W[k+1][c]))
__global__ __launch_bounds__(256) void k_packw(const float* __restrict__ W,
                                               unsigned* __restrict__ Wp, int Kd) {
  int i = blockIdx.x * 256 + threadIdx.x;
  if (i >= (Kd >> 1) * HD) return;
  int c = i & (HD - 1), kp = i >> 7;
  union { _Float16 h[2]; unsigned u; } pk;
  pk.h[0] = (_Float16)W[(2 * kp) * HD + c];
  pk.h[1] = (_Float16)W[(2 * kp + 1) * HD + c];
  Wp[i] = pk.u;
}

// ---------------------------------------------------------------- WMMA GEMM
// C[b, m, 0:128] = A[b, m, 0:KD] @ W[KD, 128]   (f16 in, f32 accum)
// KD is a template constant so the K-loop fully unrolls: one base address per
// operand, immediate-offset clause loads, back-to-back v_wmma.
template <int KD>
__global__ __launch_bounds__(256) void k_gemm(const _Float16* __restrict__ A,
                                              const unsigned* __restrict__ Wp,
                                              float* __restrict__ C) {
  const int b     = blockIdx.y;
  const int m0    = blockIdx.x << 4;
  const int wave  = threadIdx.x >> 5;
  const int lane  = threadIdx.x & 31;
  const int l16   = lane & 15;
  const int khalf = lane >> 4;                 // 0: K lo-half, 1: K hi-half
  const int row   = m0 + l16;                  // A row for this lane
  const int col   = (wave << 4) + l16;         // C/B column for this lane
  const _Float16* Arow = A + ((size_t)(b * NPAD) + row) * KD + (khalf << 3);
  const unsigned* Wcol = Wp + col;

  v8f acc = {};
#pragma unroll
  for (int k0 = 0; k0 < KD; k0 += 32) {
    v16h av, bv;
    // A: VGPRs 0-3 = K pairs k0+khalf*8 .. +7 (16 contiguous bytes),
    //    VGPRs 4-7 = same +16 halfs  -> two b128 loads
    uint4* ap = (uint4*)&av;
    ap[0] = *(const uint4*)(Arow + k0);
    ap[1] = *(const uint4*)(Arow + k0 + 16);
    // B: packed K-pair dwords, stride 128 dwords between pairs
    unsigned* bp = (unsigned*)&bv;
#pragma unroll
    for (int v = 0; v < 8; ++v) {
      int kk = k0 + ((v & 3) << 1) + (khalf << 3) + ((v >> 2) << 4);
      bp[v] = Wcol[(kk >> 1) * HD];
    }
    acc = __builtin_amdgcn_wmma_f32_16x16x32_f16(false, av, false, bv,
                                                 (short)0, acc, false, false);
  }
#pragma unroll
  for (int r = 0; r < 8; ++r) {
    int M = m0 + r + (khalf << 3);
    C[((size_t)(b * NPAD) + M) * HD + col] = acc[r];
  }
}

// ------------------------------------------------- aggregation (self + edges)
__global__ __launch_bounds__(256) void k_agginit(const float* __restrict__ Hd,
                                                 const float* __restrict__ bias,
                                                 const float* __restrict__ selfn,
                                                 float* __restrict__ Agg) {
  int i = blockIdx.x * 256 + threadIdx.x;      // over BSZ*NG*HD
  if (i >= BSZ * NG * HD) return;
  int f = i & (HD - 1);
  int nb = i >> 7;
  int n = nb % NG, b = nb / NG;
  size_t off = ((size_t)(b * NPAD) + n) * HD + f;
  Agg[off] = Hd[off] * selfn[n] + bias[f];
}

// one wave per (edge, batch); lane covers 4 features via float4
__global__ __launch_bounds__(256) void k_scatter(const float* __restrict__ Hd,
                                                 float* __restrict__ Agg,
                                                 const int* __restrict__ src,
                                                 const int* __restrict__ dst,
                                                 const float* __restrict__ en) {
  int tid  = blockIdx.x * 256 + threadIdx.x;
  int wid  = tid >> 5;
  int lane = threadIdx.x & 31;
  int e = wid >> 3;
  int b = wid & 7;
  if (e >= EDG) return;
  int s = src[e], d = dst[e];
  float w = en[e];
  const float4 v = *(const float4*)(Hd + ((size_t)(b * NPAD) + s) * HD + lane * 4);
  float* out = Agg + ((size_t)(b * NPAD) + d) * HD + lane * 4;
  unsafeAtomicAdd(out + 0, v.x * w);
  unsafeAtomicAdd(out + 1, v.y * w);
  unsafeAtomicAdd(out + 2, v.z * w);
  unsafeAtomicAdd(out + 3, v.w * w);
}

// relu + f16 copy for next layer + fold layer's slice of xc into score/proj
__global__ __launch_bounds__(256) void k_finalize(const float* __restrict__ Agg,
                                                  _Float16* __restrict__ Hout,
                                                  const float* __restrict__ tw,
                                                  const float* __restrict__ fw,
                                                  float* __restrict__ score,
                                                  float* __restrict__ proj,
                                                  int layer) {
  int tid  = blockIdx.x * 256 + threadIdx.x;
  int wid  = tid >> 5;
  int lane = threadIdx.x & 31;
  int n = wid % NPAD, b = wid / NPAD;
  if (b >= BSZ) return;
  size_t base = ((size_t)(b * NPAD) + n) * HD + lane * 4;
  if (n >= NG) {                                // keep pad rows zero for GEMM
    union { _Float16 h[4]; uint2 u; } zo; zo.u = make_uint2(0u, 0u);
    *(uint2*)(Hout + base) = zo.u;
    return;
  }
  float4 v = *(const float4*)(Agg + base);
  float r0 = fmaxf(v.x, 0.f), r1 = fmaxf(v.y, 0.f);
  float r2 = fmaxf(v.z, 0.f), r3 = fmaxf(v.w, 0.f);
  union { _Float16 h[4]; uint2 u; } o;
  o.h[0] = (_Float16)r0; o.h[1] = (_Float16)r1;
  o.h[2] = (_Float16)r2; o.h[3] = (_Float16)r3;
  *(uint2*)(Hout + base) = o.u;
  int f0 = lane * 4;                            // xc feature index = h*3+layer
  float s = r0 * tw[(f0 + 0) * 3 + layer] + r1 * tw[(f0 + 1) * 3 + layer] +
            r2 * tw[(f0 + 2) * 3 + layer] + r3 * tw[(f0 + 3) * 3 + layer];
  float p = r0 * fw[(f0 + 0) * 3 + layer] + r1 * fw[(f0 + 1) * 3 + layer] +
            r2 * fw[(f0 + 2) * 3 + layer] + r3 * fw[(f0 + 3) * 3 + layer];
  for (int o2 = 16; o2; o2 >>= 1) {
    s += __shfl_down(s, o2);
    p += __shfl_down(p, o2);
  }
  if (lane == 0) {
    score[b * NG + n] += s;
    proj[b * NG + n] += p;
  }
}

// ------------------------------- bitonic full sort in 128KB LDS, then z[b,k]
__global__ __launch_bounds__(1024) void k_sort(const float* __restrict__ score,
                                               const float* __restrict__ proj,
                                               const float* __restrict__ invn,
                                               const float* __restrict__ fcb,
                                               float* __restrict__ z) {
  extern __shared__ char smem[];
  float* ss = (float*)smem;
  int*   si = (int*)(smem + SORTN * sizeof(float));
  const int b = blockIdx.x;
  const float inv = invn[0];
  for (int i = threadIdx.x; i < SORTN; i += 1024) {
    if (i < NG) { ss[i] = tanhf(score[b * NG + i] * inv); si[i] = i; }
    else        { ss[i] = -__builtin_inff();              si[i] = 0x7fffffff; }
  }
  __syncthreads();
  // sort descending by score, ties -> smaller index first (matches lax.top_k)
  for (int k = 2; k <= SORTN; k <<= 1) {
    for (int j = k >> 1; j > 0; j >>= 1) {
      for (int i = threadIdx.x; i < SORTN; i += 1024) {
        int ix = i ^ j;
        if (ix > i) {
          float s1 = ss[i], s2 = ss[ix];
          int   i1 = si[i], i2 = si[ix];
          bool dirAsc = ((i & k) == 0);
          bool swp = dirAsc ? ((s2 > s1) || (s2 == s1 && i2 < i1))
                            : ((s1 > s2) || (s1 == s2 && i1 < i2));
          if (swp) { ss[i] = s2; ss[ix] = s1; si[i] = i2; si[ix] = i1; }
        }
      }
      __syncthreads();
    }
  }
  const float fb = fcb[0];
  for (int kq = threadIdx.x; kq < KTOP; kq += 1024)
    z[b * KTOP + kq] = ss[kq] * proj[b * NG + si[kq]] + fb;
}

// ---------------------------------------------------- lin1: [8,K]@[K,512]
__global__ __launch_bounds__(512) void k_lin1(const float* __restrict__ z,
                                              const float* __restrict__ W,
                                              float* __restrict__ h512) {
  const int hf = threadIdx.x;
  const int k0 = blockIdx.x * 237;
  const int k1 = (k0 + 237 < KTOP) ? (k0 + 237) : KTOP;
  float acc[BSZ];
#pragma unroll
  for (int b = 0; b < BSZ; ++b) acc[b] = 0.f;
  for (int k = k0; k < k1; ++k) {
    float w = W[(size_t)k * HFD + hf];
#pragma unroll
    for (int b = 0; b < BSZ; ++b) acc[b] += z[b * KTOP + k] * w;
  }
#pragma unroll
  for (int b = 0; b < BSZ; ++b) unsafeAtomicAdd(&h512[b * HFD + hf], acc[b]);
}

// ---------------------------------------------------- head + log_softmax
__global__ __launch_bounds__(32) void k_head(const float* __restrict__ h512,
                                             const float* __restrict__ l1b,
                                             const float* __restrict__ W2,
                                             const float* __restrict__ b2,
                                             float* __restrict__ out) {
  __shared__ float lg[BSZ][NC];
  int t = threadIdx.x;
  if (t < BSZ * NC) {
    int b = t >> 1, c = t & 1;
    float acc = b2[c];
    for (int hf = 0; hf < HFD; ++hf) {
      float hv = fmaxf(h512[b * HFD + hf] + l1b[hf], 0.f);
      acc += hv * W2[hf * NC + c];
    }
    lg[b][c] = acc;
  }
  __syncthreads();
  if (t < BSZ * NC) {
    int b = t >> 1, c = t & 1;
    float m = fmaxf(lg[b][0], lg[b][1]);
    float lse = m + logf(expf(lg[b][0] - m) + expf(lg[b][1] - m));
    out[t] = lg[b][c] - lse;
  }
}

// ============================================================== launch
extern "C" void kernel_launch(void* const* d_in, const int* in_sizes, int n_in,
                              void* d_out, int out_size, void* d_ws, size_t ws_size,
                              hipStream_t stream) {
  const float* x   = (const float*)d_in[0];
  const int*   ei  = (const int*)d_in[2];
  const int*   esrc = ei;
  const int*   edst = ei + EDG;
  const float* W1 = (const float*)d_in[3];  const float* b1 = (const float*)d_in[4];
  const float* W2 = (const float*)d_in[5];  const float* b2 = (const float*)d_in[6];
  const float* W3 = (const float*)d_in[7];  const float* b3 = (const float*)d_in[8];
  const float* tw = (const float*)d_in[9];
  const float* fw = (const float*)d_in[10];
  const float* fcb = (const float*)d_in[11];
  const float* l1W = (const float*)d_in[12]; const float* l1b = (const float*)d_in[13];
  const float* l2W = (const float*)d_in[14]; const float* l2b = (const float*)d_in[15];
  float* out = (float*)d_out;

  char* ws = (char*)d_ws;
  size_t off = 0;
  auto take = [&](size_t bytes) -> char* {
    char* p = ws + off;
    off = (off + bytes + 255) & ~(size_t)255;
    return p;
  };
  _Float16* xf16 = (_Float16*)take((size_t)BSZ * NPAD * FIN * 2);
  _Float16* hA   = (_Float16*)take((size_t)BSZ * NPAD * HD * 2);
  _Float16* hB   = (_Float16*)take((size_t)BSZ * NPAD * HD * 2);
  float*    Hd   = (float*)take((size_t)BSZ * NPAD * HD * 4);
  float*    Agg  = (float*)take((size_t)BSZ * NPAD * HD * 4);
  unsigned* W1p  = (unsigned*)take((size_t)(FIN / 2) * HD * 4);
  unsigned* W2p  = (unsigned*)take((size_t)(HD / 2) * HD * 4);
  unsigned* W3p  = (unsigned*)take((size_t)(HD / 2) * HD * 4);
  unsigned* deg  = (unsigned*)take((size_t)NG * 4);
  float*    dinv = (float*)take((size_t)NG * 4);
  float*    selfn= (float*)take((size_t)NG * 4);
  float*    en   = (float*)take((size_t)EDG * 4);
  float*    score= (float*)take((size_t)BSZ * NG * 4);
  float*    proj = (float*)take((size_t)BSZ * NG * 4);
  float*    invn = (float*)take(256);
  float*    z    = (float*)take((size_t)BSZ * KTOP * 4);
  float*    h512 = (float*)take((size_t)BSZ * HFD * 4);

  // setup
  k_init<<<(BSZ * NG + 255) / 256, 256, 0, stream>>>(score, proj, h512, deg);
  k_deg<<<(EDG + 255) / 256, 256, 0, stream>>>(edst, deg);
  k_dinv<<<(NG + 255) / 256, 256, 0, stream>>>(deg, dinv, selfn);
  k_enorm<<<(EDG + 255) / 256, 256, 0, stream>>>(esrc, edst, dinv, en);
  k_invnorm<<<1, 256, 0, stream>>>(tw, invn);
  k_convert_x<<<(BSZ * NPAD * FIN + 255) / 256, 256, 0, stream>>>(x, xf16);
  k_packw<<<((FIN / 2) * HD + 255) / 256, 256, 0, stream>>>(W1, W1p, FIN);
  k_packw<<<((HD / 2) * HD + 255) / 256, 256, 0, stream>>>(W2, W2p, HD);
  k_packw<<<((HD / 2) * HD + 255) / 256, 256, 0, stream>>>(W3, W3p, HD);

  const dim3 gGemm(NPAD / 16, BSZ);
  const int gAgg  = (BSZ * NG * HD + 255) / 256;
  const int gScat = EDG;                       // 8 waves/block = 8 (e,b) pairs
  const int gFin  = (BSZ * NPAD * 32 + 255) / 256;

  const _Float16* Ain[3]  = { xf16, hA, hB };
  _Float16*       Aout[3] = { hA, hB, hA };
  const unsigned* Wp[3]   = { W1p, W2p, W3p };
  const float*    bias[3] = { b1, b2, b3 };

  for (int layer = 0; layer < 3; ++layer) {
    if (layer == 0)
      k_gemm<FIN><<<gGemm, 256, 0, stream>>>(Ain[layer], Wp[layer], Hd);
    else
      k_gemm<HD><<<gGemm, 256, 0, stream>>>(Ain[layer], Wp[layer], Hd);
    k_agginit<<<gAgg, 256, 0, stream>>>(Hd, bias[layer], selfn, Agg);
    k_scatter<<<gScat, 256, 0, stream>>>(Hd, Agg, esrc, edst, en);
    k_finalize<<<gFin, 256, 0, stream>>>(Agg, Aout[layer], tw, fw, score, proj, layer);
  }

  k_sort<<<BSZ, 1024, SORTN * 8, stream>>>(score, proj, invn, fcb, z);
  k_lin1<<<32, 512, 0, stream>>>(z, l1W, h512);
  k_head<<<1, 32, 0, stream>>>(h512, l1b, l2W, l2b, out);
}
